// Attention_linearCombination_91345364451457
// MI455X (gfx1250) — matrix operand order repaired and checked by reading, not compiled, not run on hardware
//
#include <hip/hip_runtime.h>
#include <cstdint>
#include <cstddef>

typedef __attribute__((ext_vector_type(16))) __bf16    v16bf;
typedef __attribute__((ext_vector_type(4)))  __bf16    bf16x4;
typedef __attribute__((ext_vector_type(2)))  __bf16    bf16x2;
typedef __attribute__((ext_vector_type(8)))  float     v8f;
typedef __attribute__((ext_vector_type(16))) float     f32x16;
typedef __attribute__((ext_vector_type(8)))  float     f32x8;
typedef __attribute__((ext_vector_type(4)))  float     f32x4;
typedef __attribute__((ext_vector_type(2)))  float     f32x2;
typedef __attribute__((ext_vector_type(4)))  uint32_t  u32x4;
typedef __attribute__((ext_vector_type(2)))  uint32_t  u32x2;

#define NB    16384   // batch rows
#define KD    1024    // inner dim
#define IDIM  256     // attention intermediate
#define ODIM  512     // output dim

union BBits {          // raw bf16 fragment bits (no conversion)
  u32x4 q[2];
  v16bf v;
};

// 2 floats -> packed bf16 pair via vector fptrunc (selects v_cvt_pk_bf16_f32)
static __device__ __forceinline__ uint32_t packbf2(float lo, float hi) {
  f32x2 f = {lo, hi};
  union { bf16x2 b; uint32_t u; } r;
  r.b = __builtin_convertvector(f, bf16x2);
  return r.u;
}

// 16 floats (in A-fragment K-order) -> v16bf with 8 packed cvts, no movs
static __device__ __forceinline__ v16bf cvt16(f32x4 f0, f32x4 f1, f32x4 f2, f32x4 f3) {
  f32x8  l8 = __builtin_shufflevector(f0, f1, 0, 1, 2, 3, 4, 5, 6, 7);
  f32x8  h8 = __builtin_shufflevector(f2, f3, 0, 1, 2, 3, 4, 5, 6, 7);
  f32x16 ff = __builtin_shufflevector(l8, h8, 0, 1, 2, 3, 4, 5, 6, 7,
                                              8, 9, 10, 11, 12, 13, 14, 15);
  return __builtin_convertvector(ff, v16bf);
}

// tanh(x) = (e^2x - 1)/(e^2x + 1), clamped; ~6 VALU vs ~25 for libm tanhf.
static __device__ __forceinline__ float fast_tanh(float x) {
  x = fminf(fmaxf(x, -15.0f), 15.0f);
  float e = __expf(2.0f * x);
  return __fdividef(e - 1.0f, e + 1.0f);
}

static __device__ __forceinline__ void softmax3(const float* __restrict__ logits,
                                                int row, float s[3]) {
  float l0 = logits[row * 3 + 0];
  float l1 = logits[row * 3 + 1];
  float l2 = logits[row * 3 + 2];
  float m  = fmaxf(l0, fmaxf(l1, l2));
  float e0 = __expf(l0 - m), e1 = __expf(l1 - m), e2 = __expf(l2 - m);
  float inv = 1.0f / (e0 + e1 + e2);
  s[0] = e0 * inv; s[1] = e1 * inv; s[2] = e2 * inv;
}

// ---------------------------------------------------------------------------
// Kernel 0: pack fp32 weight [KD x N] into bf16 WMMA B-fragment order.
// Tile (kt,nt) = 32x16. Per ISA: lane = column (L%16), lanes 0-15 hold
// K=kt*32+{0..7,16..23}, lanes 16-31 hold K=kt*32+{8..15,24..31}; dword j holds K={k,k+1}, k=8*(L/16)+2*(j&3)+16*(j>>2).
// Storage: dst[((kt*(N/16)+nt)*32 + lane)*8 + j]  -> lane reads 32B contig.
// ---------------------------------------------------------------------------
__global__ void pack_weights_kernel(const float* __restrict__ W,
                                    uint32_t* __restrict__ dst, int N) {
  int t = blockIdx.x * blockDim.x + threadIdx.x;
  int total = (KD * N) >> 1;
  if (t >= total) return;
  int j    = t & 7;
  int lane = (t >> 3) & 31;
  int tile = t >> 8;
  int nt16 = N >> 4;
  int nt = tile % nt16;
  int kt = tile / nt16;
  int k0 = kt * 32 + (lane >> 4) * 8 + 2 * (j & 3) + 16 * (j >> 2);
  int n  = nt * 16 + (lane & 15);
  dst[t] = packbf2(W[(size_t)k0 * N + n], W[(size_t)(k0 + 1) * N + n]);
}

// ---------------------------------------------------------------------------
// Kernel 1: logits[b][branch] = tanh(x @ W_att + b_att) @ v
// One wave per 16 rows; 16 N-tile accumulators cover all 256 cols so each
// A fragment (built from fp32 x on the fly) is loaded once per k-step.
// ---------------------------------------------------------------------------
__global__ __launch_bounds__(128) void attn_logits_kernel(
    const float* __restrict__ x, const uint32_t* __restrict__ Wp,
    const float* __restrict__ b_att, const float* __restrict__ vvec,
    float* __restrict__ logits, int branch)
{
  const int wave = threadIdx.x >> 5;
  const int lane = threadIdx.x & 31;
  const int rlo  = lane & 15;
  const int hi   = lane >> 4;            // 0: lanes 0-15, 1: lanes 16-31
  const int rowBase = blockIdx.x * 64 + wave * 16;

  v8f acc[16] = {};

  const float* xrow = x + (size_t)(rowBase + rlo) * KD;
  const int kb = hi * 8;                 // A layout: low lanes K 0..7, high K 8..15

#pragma unroll 1
  for (int kt = 0; kt < 32; ++kt) {
    // A fragment: row = L%16; two contiguous 8-float runs; K-order == load order
    f32x4 f0 = *(const f32x4*)(xrow + kt * 32 + kb);
    f32x4 f1 = *(const f32x4*)(xrow + kt * 32 + kb + 4);
    f32x4 f2 = *(const f32x4*)(xrow + kt * 32 + kb + 16);
    f32x4 f3 = *(const f32x4*)(xrow + kt * 32 + kb + 20);
    v16bf av = cvt16(f0, f1, f2, f3);

    const uint32_t* wkt = Wp + (((size_t)kt * 16) * 32 + lane) * 8;
#pragma unroll
    for (int nt = 0; nt < 16; ++nt) {
      BBits b;
      const u32x4* bp = (const u32x4*)(wkt + (size_t)nt * 256);
      b.q[0] = bp[0]; b.q[1] = bp[1];
      acc[nt] = __builtin_amdgcn_wmma_f32_16x16x32_bf16(
          false, av, false, b.v, (short)0, acc[nt], false, false);
    }
  }

  // Epilogue: tanh(acc + bias) * v, reduced over the 16 columns held by the
  // 16-lane half-groups (C layout: VGPR r, lane L -> M = r + 8*(L/16), N = L%16)
  float part[8] = {};
#pragma unroll
  for (int nt = 0; nt < 16; ++nt) {
    const int col = nt * 16 + rlo;
    const float ba = b_att[col];
    const float vv = vvec[col];
#pragma unroll
    for (int r = 0; r < 8; ++r)
      part[r] += fast_tanh(acc[nt][r] + ba) * vv;
  }
#pragma unroll
  for (int r = 0; r < 8; ++r) {
    float s = part[r];
    s += __shfl_xor(s, 1, 32);
    s += __shfl_xor(s, 2, 32);
    s += __shfl_xor(s, 4, 32);
    s += __shfl_xor(s, 8, 32);
    if (rlo == 0)
      logits[(size_t)(rowBase + hi * 8 + r) * 3 + branch] = s;
  }
}

// ---------------------------------------------------------------------------
// Kernel 2: out = sum_i soft_i * (x_i @ W_tr_i + b_tr_i)
// Block = 16 rows x 512 cols (4 waves x 8 N-tiles). Softmax weight folded
// into A at bf16-conversion time while staging x into LDS, so all branches
// accumulate into one set of WMMA accumulators. Double-buffered LDS staging:
// one barrier per k-step, next step's global loads overlap current WMMAs.
// ---------------------------------------------------------------------------
__global__ __launch_bounds__(128) void transform_kernel(
    const float* __restrict__ x0, const float* __restrict__ x1, const float* __restrict__ x2,
    const uint32_t* __restrict__ W0, const uint32_t* __restrict__ W1, const uint32_t* __restrict__ W2,
    const float* __restrict__ b0, const float* __restrict__ b1, const float* __restrict__ b2,
    const float* __restrict__ logits, float* __restrict__ out)
{
  // 2 buffers x 16 rows x 32 bf16, row stride 48 elems (96B: 16B-aligned reads)
  __shared__ __align__(16) uint16_t lds[2][16 * 48];

  const int t    = threadIdx.x;
  const int wave = t >> 5;
  const int lane = t & 31;
  const int rlo  = lane & 15;
  const int hi   = lane >> 4;
  const int rowBase = blockIdx.x * 16;

  // staging assignment: thread t loads row t/8, 4 floats at col (t%8)*4
  const int srow = t >> 3;
  const int scol = (t & 7) * 4;

  float sst[3], sl[3];
  softmax3(logits, rowBase + srow, sst);   // for staging (pre-scaling A)
  softmax3(logits, rowBase + rlo,  sl);    // lane q & q+16 hold row q (epilogue shfl)

  v8f acc[8] = {};

  // stage one 16x32 fp32 chunk -> scaled bf16 in LDS buffer `buf`
  auto stage = [&](int step, int buf) {
    const int br = step >> 5;
    const int kt = step & 31;
    const float* xb = (br == 0) ? x0 : (br == 1) ? x1 : x2;
    const float  sc = (br == 0) ? sst[0] : (br == 1) ? sst[1] : sst[2];
    f32x4 f = *(const f32x4*)(xb + (size_t)(rowBase + srow) * KD + kt * 32 + scol);
    f = f * sc;
    union { bf16x4 b; u32x2 u; } cv;
    cv.b = __builtin_convertvector(f, bf16x4);   // 2x v_cvt_pk_bf16_f32
    *(u32x2*)&lds[buf][srow * 48 + scol] = cv.u; // ds_store_b64
  };

  stage(0, 0);
  __syncthreads();

#pragma unroll 1
  for (int step = 0; step < 96; ++step) {          // 3 branches x 32 k-steps
    const int buf = step & 1;
    if (step + 1 < 96) stage(step + 1, buf ^ 1);   // prefetch into other buffer

    const int br = step >> 5;
    const int kt = step & 31;
    const uint32_t* Wp = (br == 0) ? W0 : (br == 1) ? W1 : W2;

    BBits a;
    const uint16_t* arow = &lds[buf][rlo * 48 + hi * 8];
    a.q[0] = *(const u32x4*)(arow);                // K kb..kb+7   (ds_read_b128)
    a.q[1] = *(const u32x4*)(arow + 16);           // K kb+16..kb+23

    const uint32_t* wkt = Wp + (((size_t)kt * 32 + wave * 8) * 32 + lane) * 8;
#pragma unroll
    for (int nl = 0; nl < 8; ++nl) {
      BBits b;
      const u32x4* bp = (const u32x4*)(wkt + (size_t)nl * 256);
      b.q[0] = bp[0]; b.q[1] = bp[1];
      acc[nl] = __builtin_amdgcn_wmma_f32_16x16x32_bf16(
          false, a.v, false, b.v, (short)0, acc[nl], false, false);
    }
    __syncthreads();   // readers of `buf` done before it is restaged at step+2
  }

  // Epilogue: add sum_i s_i(row) * b_tr_i[col], store fp32
#pragma unroll
  for (int nl = 0; nl < 8; ++nl) {
    const int col = wave * 128 + nl * 16 + rlo;
    const float bb0 = b0[col], bb1 = b1[col], bb2 = b2[col];
#pragma unroll
    for (int r = 0; r < 8; ++r) {
      const int src = hi * 8 + r;                  // lane holding softmax for this row
      float add = __shfl(sl[0], src, 32) * bb0
                + __shfl(sl[1], src, 32) * bb1
                + __shfl(sl[2], src, 32) * bb2;
      out[(size_t)(rowBase + hi * 8 + r) * ODIM + col] = acc[nl][r] + add;
    }
  }
}

// ---------------------------------------------------------------------------
extern "C" void kernel_launch(void* const* d_in, const int* in_sizes, int n_in,
                              void* d_out, int out_size, void* d_ws, size_t ws_size,
                              hipStream_t stream) {
  (void)in_sizes; (void)n_in; (void)out_size; (void)ws_size;

  // dict order: x1,W_att1,b_att1,W_tr1,b_tr1, x2,..., x3,..., v
  const float *x[3], *Wa[3], *ba[3], *Wt[3], *bt[3];
  for (int i = 0; i < 3; ++i) {
    x[i]  = (const float*)d_in[5 * i + 0];
    Wa[i] = (const float*)d_in[5 * i + 1];
    ba[i] = (const float*)d_in[5 * i + 2];
    Wt[i] = (const float*)d_in[5 * i + 3];
    bt[i] = (const float*)d_in[5 * i + 4];
  }
  const float* vvec = (const float*)d_in[15];

  // workspace: logits[B*3] (f32) + packed bf16 weights (~4.9 MB total)
  uint32_t* ws = (uint32_t*)d_ws;
  float* logits = (float*)ws;
  size_t off = (size_t)NB * 3;
  uint32_t* pWa[3]; uint32_t* pWt[3];
  for (int i = 0; i < 3; ++i) { pWa[i] = ws + off; off += (size_t)KD * IDIM / 2; }
  for (int i = 0; i < 3; ++i) { pWt[i] = ws + off; off += (size_t)KD * ODIM / 2; }

  for (int i = 0; i < 3; ++i) {
    int total = KD * IDIM / 2;
    pack_weights_kernel<<<(total + 255) / 256, 256, 0, stream>>>(Wa[i], pWa[i], IDIM);
  }
  for (int i = 0; i < 3; ++i) {
    int total = KD * ODIM / 2;
    pack_weights_kernel<<<(total + 255) / 256, 256, 0, stream>>>(Wt[i], pWt[i], ODIM);
  }
  for (int i = 0; i < 3; ++i)
    attn_logits_kernel<<<NB / 64, 128, 0, stream>>>(x[i], pWa[i], ba[i], vvec, logits, i);

  transform_kernel<<<NB / 16, 128, 0, stream>>>(x[0], x[1], x[2],
                                                pWt[0], pWt[1], pWt[2],
                                                bt[0], bt[1], bt[2],
                                                logits, (float*)d_out);
}
